// ScaledDotProductAttention_bias_9732395892963
// MI455X (gfx1250) — matrix operand order repaired and checked by reading, not compiled, not run on hardware
//
#include <hip/hip_runtime.h>
#include <hip/hip_bf16.h>

// Problem constants (match reference)
#define BB 2
#define HH 16
#define SSQ 1024   // seq len
#define DH 64      // head dim
#define QB 64      // query rows per block
#define NT 128     // 4 waves of 32
#define NKT (SSQ / 16)
#define NEGV (-1.0e9f)

// ---- LDS layout (bytes) ----
// sc      : QB*SSQ f32                  = 262144  @ 0
// rowInv  : QB f32 (+pad)               =    512  @ 262144
// kstage  : 2 x (16*DH) f32 (dbl-buf)   =   8192  @ 262656
// qtile   : QB*DH bf16                  =   8192  @ 270848
// vtstage : 64*34 bf16 (V^T chunk)      =   4352  @ 279040
#define SC_ELEMS   (QB * SSQ)
#define KSTAGE_OFF 262656u
#define SMEM_BYTES 283392

typedef __bf16 v16bf  __attribute__((ext_vector_type(16)));
typedef __bf16 bf16x2 __attribute__((ext_vector_type(2)));
typedef float  v8f    __attribute__((ext_vector_type(8)));
typedef float  f32x4  __attribute__((ext_vector_type(4)));

#if defined(__has_builtin)
#if __has_builtin(__builtin_amdgcn_tensor_load_to_lds) && \
    __has_builtin(__builtin_amdgcn_s_wait_tensorcnt)
#define USE_TDM 1
#else
#define USE_TDM 0
#endif
#else
#define USE_TDM 0
#endif

// Native converts (v_cvt_*bf16*) instead of software RNE bit-twiddling.
__device__ __forceinline__ __bf16 f2bf(float f) { return (__bf16)f; }
__device__ __forceinline__ unsigned pack2bf(float a, float b) {
  bf16x2 v; v[0] = (__bf16)a; v[1] = (__bf16)b;
  unsigned u; __builtin_memcpy(&u, &v, 4); return u;
}

__device__ __forceinline__ v8f wmma_bf16(v16bf a, v16bf b, v8f c) {
  return __builtin_amdgcn_wmma_f32_16x16x32_bf16(
      false, a, false, b, (short)0, c, false, false);
}

#if USE_TDM
typedef unsigned int u32x4 __attribute__((ext_vector_type(4)));
typedef int          i32x8 __attribute__((ext_vector_type(8)));
typedef int          i32x4 __attribute__((ext_vector_type(4)));

// 1-D TDM copy: 1024 f32 (one 16x64 K tile, contiguous) global -> LDS.
__device__ __forceinline__ void tdm_load_ktile(const float* gsrc, unsigned lds_off) {
  unsigned long long ga = (unsigned long long)gsrc;
  u32x4 g0;
  g0[0] = 1u;                                   // count=1 (valid descriptor)
  g0[1] = lds_off;                              // lds_addr (bytes)
  g0[2] = (unsigned)ga;                         // global_addr[31:0]
  g0[3] = ((unsigned)(ga >> 32) & 0x1FFFFFFu)   // global_addr[56:32]
          | (2u << 30);                         // type=2 ("image")
  i32x8 g1;
  g1[0] = 0x20000;        // data_size=2 (4B); no multicast/iterate/pad
  g1[1] = 0x04000000;     // tensor_dim0 = 1024  (bits 79:48)
  g1[2] = 0x00010000;     // tensor_dim1 = 1     (bits 111:80)
  g1[3] = 0x04000000;     // tile_dim0   = 1024  (bits 127:112)
  g1[4] = 1;              // tile_dim1 = 1, tile_dim2 = 0
  g1[5] = 1024;           // tensor_dim0_stride (unused: single row)
  g1[6] = 0;
  g1[7] = 0;
  i32x4 gz4 = {0, 0, 0, 0};
  i32x8 gz8 = {0, 0, 0, 0, 0, 0, 0, 0};
  // 6-arg form on this toolchain: (g0, g1, g2, g3, extra, cpol)
  __builtin_amdgcn_tensor_load_to_lds(g0, g1, gz4, gz4, gz8, 0);
}
#endif

__global__ void __launch_bounds__(NT, 1)
fused_attn_bias_softmax_pv(const float* __restrict__ Q,
                           const float* __restrict__ K,
                           const float* __restrict__ V,
                           const unsigned char* __restrict__ mask,
                           const float* __restrict__ ia_t,
                           const float* __restrict__ ia_f,
                           float* __restrict__ ctx_out,
                           float* __restrict__ attn_out) {
  extern __shared__ char smem_raw[];
  float*  sc     = (float*)smem_raw;                    // [QB][SSQ]
  float*  rowInv = (float*)(smem_raw + SC_ELEMS * 4);   // [QB]
  float*  kst0   = (float*)(smem_raw + KSTAGE_OFF);     // 2 x [16][DH] f32
  __bf16* qtile  = (__bf16*)(smem_raw + 270848);        // [QB][DH]
  __bf16* vtst   = (__bf16*)(smem_raw + 279040);        // [64][34] V^T bf16

  const int tid   = threadIdx.x;
  const int lane  = tid & 31;
  const int mrow  = lane & 15;
  const int lhalf = lane >> 4;
  const int m0    = (tid >> 5) * 16;   // wave's query rows

  const int bh = blockIdx.x >> 4;
  const int q0 = (blockIdx.x & 15) * QB;
  const size_t qkvOff  = (size_t)bh * SSQ * DH;
  const size_t attnOff = (size_t)bh * SSQ * SSQ + (size_t)q0 * SSQ;
  const float* Qb = Q + qkvOff + (size_t)q0 * DH;
  const float* Kb = K + qkvOff;
  const float* Vb = V + qkvOff;

  // ---- P0: Q slab -> LDS bf16 (b128 loads, packed-cvt uint2 stores) ----
  {
    const f32x4* Qv = (const f32x4*)Qb;
    uint2* qv = (uint2*)qtile;
    for (int i = tid; i < QB * DH / 4; i += NT) {
      const f32x4 q = Qv[i];
      qv[i] = make_uint2(pack2bf(q.x, q.y), pack2bf(q.z, q.w));
    }
  }
#if USE_TDM
  if (tid < 32) tdm_load_ktile(Kb, KSTAGE_OFF);   // prime the pipeline
#endif
  __syncthreads();

  // ---- P1: scores = Q K^T  (TDM double-buffered K tiles) ----
  for (int kt = 0; kt < NKT; ++kt) {
    float* kst = kst0 + (kt & 1) * (16 * DH);
#if USE_TDM
    if (tid < 32) {
      if (kt + 1 < NKT) {
        tdm_load_ktile(Kb + (size_t)(kt + 1) * 16 * DH,
                       KSTAGE_OFF + (unsigned)((kt + 1) & 1) * 4096u);
        __builtin_amdgcn_s_wait_tensorcnt(1);   // buf[kt] complete
      } else {
        __builtin_amdgcn_s_wait_tensorcnt(0);
      }
    }
#else
    {
      const f32x4* Ks = (const f32x4*)(Kb + (size_t)kt * 16 * DH);
      f32x4* kd = (f32x4*)kst;
      for (int i = tid; i < 16 * DH / 4; i += NT) kd[i] = Ks[i];
      if (kt + 1 < NKT && tid < 16)
        __builtin_prefetch(Kb + (size_t)(kt + 1) * 16 * DH + tid * DH, 0, 1);
    }
#endif
    __syncthreads();   // publish buf[kt] to all waves

    v8f acc;
#pragma unroll
    for (int r = 0; r < 8; ++r) acc[r] = 0.f;
#pragma unroll
    for (int c = 0; c < 2; ++c) {
      v16bf a, b;
#pragma unroll
      for (int j = 0; j < 16; ++j) {
        const int da = c * 32 + lhalf * 16 + j;
        a[j] = qtile[(m0 + mrow) * DH + da];
        b[j] = f2bf(kst[mrow * DH + c * 32 + lhalf * 16 + j]);
      }
      acc = wmma_bf16(a, b, acc);
    }
#pragma unroll
    for (int r = 0; r < 8; ++r)
      sc[(m0 + lhalf * 8 + r) * SSQ + kt * 16 + mrow] = acc[r];
    __syncthreads();   // all waves done with buf[kt] before TDM overwrites it
  }

  // ---- P2: bias+mask sweep: b128 non-temporal streams + one b32 mask word ----
  {
    const f32x4* t4p = (const f32x4*)(ia_t + attnOff);
    const f32x4* f4p = (const f32x4*)(ia_f + attnOff);
    const unsigned* m4p = (const unsigned*)(mask + attnOff);
    f32x4* s4p = (f32x4*)sc;
    for (int i = tid; i < QB * SSQ / 4; i += NT) {
      const f32x4 t4 = __builtin_nontemporal_load(&t4p[i]);
      const f32x4 f4 = __builtin_nontemporal_load(&f4p[i]);
      const unsigned m4 = __builtin_nontemporal_load(&m4p[i]);
      const f32x4 s4 = s4p[i];
      f32x4 r;
      r.x = (m4 & 0x000000FFu) ? NEGV : fmaf(s4.x, 0.125f, t4.x + f4.x);
      r.y = (m4 & 0x0000FF00u) ? NEGV : fmaf(s4.y, 0.125f, t4.y + f4.y);
      r.z = (m4 & 0x00FF0000u) ? NEGV : fmaf(s4.z, 0.125f, t4.z + f4.z);
      r.w = (m4 & 0xFF000000u) ? NEGV : fmaf(s4.w, 0.125f, t4.w + f4.w);
      s4p[i] = r;
    }
  }
  __syncthreads();

  // ---- P3: row softmax stats (2 lanes/row, b128 LDS sweeps) ----
  {
    const int row  = tid >> 1;
    const int half = tid & 1;
    f32x4* rp = (f32x4*)(sc + row * SSQ + half * (SSQ / 2));
    float m = -3.0e38f;
    for (int k = 0; k < SSQ / 8; ++k) {
      const f32x4 v = rp[k];
      m = fmaxf(m, fmaxf(fmaxf(v.x, v.y), fmaxf(v.z, v.w)));
    }
    m = fmaxf(m, __shfl_xor(m, 1, 32));
    float l = 0.f;
    for (int k = 0; k < SSQ / 8; ++k) {
      f32x4 v = rp[k];
      v.x = __expf(v.x - m); v.y = __expf(v.y - m);
      v.z = __expf(v.z - m); v.w = __expf(v.w - m);
      rp[k] = v;
      l += (v.x + v.y) + (v.z + v.w);
    }
    l += __shfl_xor(l, 1, 32);
    if (half == 0) rowInv[row] = 1.0f / l;
  }
  __syncthreads();

  // ---- P4: normalize, stream attn to HBM (non-temporal b128) ----
  {
    f32x4* s4p = (f32x4*)sc;
    f32x4* o4p = (f32x4*)(attn_out + attnOff);
    for (int i = tid; i < QB * SSQ / 4; i += NT) {
      const float rinv = rowInv[i >> 8];   // 256 f32x4 per row
      f32x4 p = s4p[i];
      p.x *= rinv; p.y *= rinv; p.z *= rinv; p.w *= rinv;
      __builtin_nontemporal_store(p, &o4p[i]);
      s4p[i] = p;
    }
  }
  __syncthreads();

  // ---- P5: context = attn * V ----
  v8f acc4[4];
#pragma unroll
  for (int nt = 0; nt < 4; ++nt)
#pragma unroll
    for (int r = 0; r < 8; ++r) acc4[nt][r] = 0.f;

  for (int vc = 0; vc < SSQ / 32; ++vc) {
    // Stage V chunk transposed as bf16: two k-neighbors packed per b32 store.
    const f32x4* Vs = (const f32x4*)(Vb + (size_t)vc * 32 * DH);
    for (int it = tid; it < 32 * DH / 8; it += NT) {  // (k-pair, d-quad) items
      const int kp = it >> 4;        // k = 2*kp
      const int dq = it & 15;        // d0 = dq*4
      const f32x4 va = Vs[(2 * kp + 0) * (DH / 4) + dq];
      const f32x4 vb = Vs[(2 * kp + 1) * (DH / 4) + dq];
      unsigned* dst0 = (unsigned*)(vtst + (dq * 4 + 0) * 34 + 2 * kp);
      unsigned* dst1 = (unsigned*)(vtst + (dq * 4 + 1) * 34 + 2 * kp);
      unsigned* dst2 = (unsigned*)(vtst + (dq * 4 + 2) * 34 + 2 * kp);
      unsigned* dst3 = (unsigned*)(vtst + (dq * 4 + 3) * 34 + 2 * kp);
      *dst0 = pack2bf(va.x, vb.x);
      *dst1 = pack2bf(va.y, vb.y);
      *dst2 = pack2bf(va.z, vb.z);
      *dst3 = pack2bf(va.w, vb.w);
    }
    if (vc + 1 < SSQ / 32 && tid < 32)
      __builtin_prefetch(Vb + (size_t)(vc + 1) * 32 * DH + tid * DH, 0, 1);
    __syncthreads();

    v16bf a;
#pragma unroll
    for (int j = 0; j < 16; ++j) {
      const int kk = vc * 32 + lhalf * 16 + j;
      a[j] = f2bf(sc[(m0 + mrow) * SSQ + kk]);
    }
#pragma unroll
    for (int nt = 0; nt < 4; ++nt) {
      v16bf b;
#pragma unroll
      for (int j = 0; j < 16; ++j)
        b[j] = vtst[(nt * 16 + mrow) * 34 + lhalf * 16 + j];
      acc4[nt] = wmma_bf16(a, b, acc4[nt]);
    }
    __syncthreads();
  }

  // ---- P6: write context ----
#pragma unroll
  for (int nt = 0; nt < 4; ++nt)
#pragma unroll
    for (int r = 0; r < 8; ++r)
      ctx_out[qkvOff + (size_t)(q0 + m0 + lhalf * 8 + r) * DH + nt * 16 + mrow] =
          acc4[nt][r];
}

extern "C" void kernel_launch(void* const* d_in, const int* in_sizes, int n_in,
                              void* d_out, int out_size, void* d_ws, size_t ws_size,
                              hipStream_t stream) {
  // inputs: Q, K, V, attn_mask(unused), attn_mask_if, ia_type, ia_feat,
  //         distance_matrix(unused by reference)
  const float* Q = (const float*)d_in[0];
  const float* K = (const float*)d_in[1];
  const float* V = (const float*)d_in[2];
  const unsigned char* mask = (const unsigned char*)d_in[4];
  const float* ia_t = (const float*)d_in[5];
  const float* ia_f = (const float*)d_in[6];

  float* ctx  = (float*)d_out;                               // (B,H,S,D)
  float* attn = (float*)d_out + (size_t)BB * HH * SSQ * DH;  // (B,H,S,S)

  (void)in_sizes; (void)n_in; (void)out_size; (void)d_ws; (void)ws_size;

  (void)hipFuncSetAttribute((const void*)fused_attn_bias_softmax_pv,
                            hipFuncAttributeMaxDynamicSharedMemorySize,
                            SMEM_BYTES);

  const dim3 grid(BB * HH * (SSQ / QB));   // 512 blocks
  fused_attn_bias_softmax_pv<<<grid, NT, SMEM_BYTES, stream>>>(
      Q, K, V, mask, ia_t, ia_f, ctx, attn);
}
